// SimpleRNN_42434276884599
// MI455X (gfx1250) — matrix operand order repaired and checked by reading, not compiled, not run on hardware
//
#include <hip/hip_runtime.h>
#include <hip/hip_bf16.h>
#include <stdint.h>

// Problem dims (fixed by the reference).
#define B_   8
#define T_   512
#define V_   32000
#define E_   512
#define H_   1024
#define MTOK (B_ * T_)   // 4096 tokens
#define G3H  (3 * H_)    // 3072 gate rows

typedef __attribute__((ext_vector_type(16))) __bf16        v16bf;
typedef __attribute__((ext_vector_type(8)))  float          v8f;
typedef __attribute__((ext_vector_type(8)))  unsigned short u16x8;
typedef int v4i __attribute__((vector_size(16)));   // matches async-LDS builtin param

#define AS1 __attribute__((address_space(1)))
#define AS3 __attribute__((address_space(3)))

#if defined(__has_builtin)
#if __has_builtin(__builtin_amdgcn_global_load_async_to_lds_b128)
#define HAVE_ASYNC_LDS 1
#endif
#endif

union FragAB { v16bf bf; u16x8 h[2]; };

__device__ __forceinline__ unsigned short f2bf(float f) {
  unsigned int u = __float_as_uint(f);
  unsigned int r = u + 0x7FFFu + ((u >> 16) & 1u);     // round-to-nearest-even
  if ((u & 0x7F800000u) == 0x7F800000u) r = u;          // Inf/NaN passthrough
  return (unsigned short)(r >> 16);
}

__device__ __forceinline__ float sigmoidf_(float x) {
  return 1.0f / (1.0f + __expf(-x));
}

// ---------------- f32 -> bf16 weight cast ----------------
// Source is read exactly once -> non-temporal loads keep L2 for the hot
// bf16 working set (w_hh / fc_w copies) instead of dead fp32 weights.
__global__ void cast_bf16_kernel(const float* __restrict__ src,
                                 unsigned short* __restrict__ dst, int n) {
  int i = blockIdx.x * blockDim.x + threadIdx.x;
  if (i < n) dst[i] = f2bf(__builtin_nontemporal_load(&src[i]));
}

// ---------------- embedding gather + cast ----------------
__global__ void embed_bf16_kernel(const long long* __restrict__ x,
                                  const float* __restrict__ emb,
                                  unsigned short* __restrict__ e_bf) {
  int i = blockIdx.x * blockDim.x + threadIdx.x;        // over MTOK*E_
  if (i >= MTOK * E_) return;
  int tok = i >> 9;                                      // / E_ (E_=512)
  int k   = i & (E_ - 1);
  long long row = x[tok];
  // tokens can repeat -> regular (RT) loads for emb rows
  e_bf[i] = f2bf(emb[row * (long long)E_ + k]);
}

// ---------------- WMMA GEMM: C[M,N] = A[M,K] * W[N,K]^T + bias ----------------
// A,W bf16 row-major; C fp32 row-major. 128x128 block tile, 8 waves of 32x64.
// Requires M%128==0, N%128==0, K%32==0 (true for all uses here).
// NT=true: stream the output with non-temporal stores (logits are write-once).
template <bool NT>
__global__ __launch_bounds__(256)
void gemm_bf16_f32_kernel(const unsigned short* __restrict__ A,
                          const unsigned short* __restrict__ W,
                          const float* __restrict__ bias,
                          float* __restrict__ C,
                          int M, int N, int K) {
  const int lane   = threadIdx.x & 31;
  const int wave   = threadIdx.x >> 5;
  const int wr     = wave >> 1;          // 0..3 : wave row group
  const int wc     = wave & 1;           // 0..1 : wave col group
  const int lane_m = lane & 15;
  const int hiHalf = lane >> 4;          // 0 or 1 (K half select per ISA layout)

  const int rowBase = blockIdx.y * 128 + wr * 32;
  const int colBase = blockIdx.x * 128 + wc * 64;

  v8f acc[2][4];
  #pragma unroll
  for (int i = 0; i < 2; ++i)
    #pragma unroll
    for (int j = 0; j < 4; ++j) acc[i][j] = {};

  for (int k0 = 0; k0 < K; k0 += 32) {
    FragAB a[2], b[4];
    // A fragment (16x32 bf16): lanes 0-15 hold K {0..7,16..23}, lanes 16-31 K {8..15,24..31}
    #pragma unroll
    for (int i = 0; i < 2; ++i) {
      const unsigned short* ap =
          A + (size_t)(rowBase + i * 16 + lane_m) * K + k0 + hiHalf * 8;
      a[i].h[0] = *(const u16x8*)(ap);
      a[i].h[1] = *(const u16x8*)(ap + 16);
    }
    // B fragment (32x16 bf16): lanes 0-15 hold K {0..7,16..23}, lanes 16-31 K {8..15,24..31}
    #pragma unroll
    for (int j = 0; j < 4; ++j) {
      const unsigned short* bp =
          W + (size_t)(colBase + j * 16 + lane_m) * K + k0 + hiHalf * 8;
      b[j].h[0] = *(const u16x8*)(bp);
      b[j].h[1] = *(const u16x8*)(bp + 16);
    }
    #pragma unroll
    for (int i = 0; i < 2; ++i)
      #pragma unroll
      for (int j = 0; j < 4; ++j)
        acc[i][j] = __builtin_amdgcn_wmma_f32_16x16x32_bf16(
            false, a[i].bf, false, b[j].bf, (short)0, acc[i][j], false, false);
  }

  // D layout: VGPR v -> row (v + 8*hiHalf), col lane_m within the 16x16 tile.
  #pragma unroll
  for (int i = 0; i < 2; ++i) {
    #pragma unroll
    for (int j = 0; j < 4; ++j) {
      int col = colBase + j * 16 + lane_m;
      float bv = bias[col];
      #pragma unroll
      for (int v = 0; v < 8; ++v) {
        int row = rowBase + i * 16 + v + 8 * hiHalf;
        float val = acc[i][j][v] + bv;
        if (NT) {
          __builtin_nontemporal_store(val, &C[(size_t)row * N + col]);
        } else {
          C[(size_t)row * N + col] = val;
        }
      }
    }
  }
}

// ---------------- persistent GRU recurrence ----------------
// Grid = 64 WGs x 96 threads (3 waves, one per gate). WG b owns hidden units
// [16b, 16b+16). Each step:
//   1) stage h (8 x 1024 bf16) global->LDS. Preferred: CDNA5 async copy
//      (GLOBAL_LOAD_ASYNC_TO_LDS_B128, ASYNCcnt-tracked, no VGPR round-trip,
//      all copies in flight at once) + s_wait_asynccnt. Fallback: batched
//      loads-then-stores so the global loads issue as one clause.
//   2) gh = h @ w_hh^T: per-gate chained WMMA column fed by ds_load_b128
//      (w_hh B-fragments are loop-invariant; compiler pins them in VGPRs),
//   3) LDS gate exchange + wave-0 fused gate math,
//   4) device-scope grid barrier.
// LDS h tile: 16 rows (8..15 zeroed once = WMMA padding), rows padded by 8
// elements so fragment reads are bank-conflict-optimal.
#define GRU_WGS 64
#define HPAD (H_ + 8)
#define NSTG 11   // ceil(8*128 / 96) staging chunks per thread
__global__ __launch_bounds__(96)
void gru_persistent_kernel(const float* __restrict__ gx,          // [MTOK, 3H] fp32
                           const unsigned short* __restrict__ w_hh,// [3H, H] bf16
                           const float* __restrict__ b_hh,        // [3H]
                           unsigned short* __restrict__ seq_out,  // [MTOK, H] bf16
                           float* __restrict__ h_f32,             // [B_, H] state
                           unsigned short* __restrict__ h_bf,     // [B_, H] state
                           unsigned int* bar) {
  const int lane   = threadIdx.x & 31;
  const int gate   = threadIdx.x >> 5;   // 0=r, 1=z, 2=n
  const int lane_m = lane & 15;
  const int hiHalf = lane >> 4;
  const int j0     = blockIdx.x * 16;

  __shared__ __attribute__((aligned(16))) unsigned short h_s[16][HPAD];
  __shared__ float gh_s[3][B_][16];

  const unsigned short* wbase = w_hh + (size_t)(gate * H_ + j0 + lane_m) * H_;
  const u16x8 zv = {0, 0, 0, 0, 0, 0, 0, 0};

  // zero the padding rows 8..15 once (8 rows x 128 chunks of 8 elems)
  for (int i = threadIdx.x; i < 8 * 128; i += 96) {
    int r = 8 + (i >> 7), c = (i & 127) * 8;
    *(u16x8*)&h_s[r][c] = zv;
  }

  for (int t = 0; t < T_; ++t) {
    // ---- stage h global -> LDS (rows 0..7) ----
#if defined(HAVE_ASYNC_LDS)
    #pragma unroll
    for (int u = 0; u < NSTG; ++u) {
      int i = (int)threadIdx.x + u * 96;
      if (i < 8 * 128) {
        int r = i >> 7, c = (i & 127) * 8;
        __builtin_amdgcn_global_load_async_to_lds_b128(
            (AS1 v4i*)(h_bf + r * H_ + c),
            (AS3 v4i*)&h_s[r][c], 0, 0);
      }
    }
    asm volatile("s_wait_asynccnt 0x0" ::: "memory");
#else
    {
      u16x8 tmp[NSTG];
      #pragma unroll
      for (int u = 0; u < NSTG; ++u) {
        int i = (int)threadIdx.x + u * 96;
        if (i < 8 * 128) {
          int r = i >> 7, c = (i & 127) * 8;
          tmp[u] = *(const u16x8*)(h_bf + r * H_ + c);
        }
      }
      #pragma unroll
      for (int u = 0; u < NSTG; ++u) {
        int i = (int)threadIdx.x + u * 96;
        if (i < 8 * 128) {
          int r = i >> 7, c = (i & 127) * 8;
          *(u16x8*)&h_s[r][c] = tmp[u];
        }
      }
    }
#endif
    __syncthreads();

    v8f acc = {};
    for (int k0 = 0; k0 < H_; k0 += 32) {
      FragAB a, b;
      const unsigned short* ap = &h_s[lane_m][k0 + hiHalf * 8];
      a.h[0] = *(const u16x8*)(ap);
      a.h[1] = *(const u16x8*)(ap + 16);
      const unsigned short* bp = wbase + k0 + hiHalf * 8;
      b.h[0] = *(const u16x8*)(bp);
      b.h[1] = *(const u16x8*)(bp + 16);
      acc = __builtin_amdgcn_wmma_f32_16x16x32_bf16(
          false, a.bf, false, b.bf, (short)0, acc, false, false);
    }
    if (hiHalf == 0) {                   // lanes 0-15 / VGPR v hold batch row v
      #pragma unroll
      for (int v = 0; v < 8; ++v) gh_s[gate][v][lane_m] = acc[v];
    }
    __syncthreads();

    if (gate == 0) {                     // wave 0: 128 (b,j) cells in 4 rounds
      #pragma unroll
      for (int rnd = 0; rnd < 4; ++rnd) {
        int idx = rnd * 32 + lane;       // 0..127
        int bb  = idx >> 4;
        int j   = idx & 15;
        int jj  = j0 + j;
        size_t gbase = ((size_t)bb * T_ + t) * G3H;
        float xr = gx[gbase + jj];
        float xz = gx[gbase + H_ + jj];
        float xn = gx[gbase + 2 * H_ + jj];
        float hr = gh_s[0][bb][j] + b_hh[jj];
        float hz = gh_s[1][bb][j] + b_hh[H_ + jj];
        float hn = gh_s[2][bb][j] + b_hh[2 * H_ + jj];
        float r = sigmoidf_(xr + hr);
        float z = sigmoidf_(xz + hz);
        float n = tanhf(xn + r * hn);
        float ho = h_f32[bb * H_ + jj];
        float hnew = (1.0f - z) * n + z * ho;
        h_f32[bb * H_ + jj] = hnew;
        unsigned short hb = f2bf(hnew);
        h_bf[bb * H_ + jj] = hb;
        seq_out[((size_t)bb * T_ + t) * H_ + jj] = hb;
      }
    }
    __syncthreads();

    // ---- grid-wide barrier: publish h, wait for all 64 WGs ----
    if (threadIdx.x == 0) {
      __builtin_amdgcn_fence(__ATOMIC_RELEASE, "agent");
      __hip_atomic_fetch_add(bar, 1u, __ATOMIC_RELAXED, __HIP_MEMORY_SCOPE_AGENT);
      unsigned target = (unsigned)gridDim.x * (unsigned)(t + 1);
      while (__hip_atomic_load(bar, __ATOMIC_RELAXED, __HIP_MEMORY_SCOPE_AGENT) < target) {
        __builtin_amdgcn_s_sleep(2);
      }
      __builtin_amdgcn_fence(__ATOMIC_ACQUIRE, "agent");
    }
    __syncthreads();
  }
}

// ---------------- host orchestration ----------------
extern "C" void kernel_launch(void* const* d_in, const int* in_sizes, int n_in,
                              void* d_out, int out_size, void* d_ws, size_t ws_size,
                              hipStream_t stream) {
  (void)in_sizes; (void)n_in; (void)out_size; (void)ws_size;

  const long long* x  = (const long long*)d_in[0];   // int64 token ids
  const float* emb    = (const float*)d_in[1];
  const float* w_ih0  = (const float*)d_in[2];
  const float* w_hh0  = (const float*)d_in[3];
  const float* b_ih0  = (const float*)d_in[4];
  const float* b_hh0  = (const float*)d_in[5];
  const float* w_ih1  = (const float*)d_in[6];
  const float* w_hh1  = (const float*)d_in[7];
  const float* b_ih1  = (const float*)d_in[8];
  const float* b_hh1  = (const float*)d_in[9];
  const float* fc_w   = (const float*)d_in[10];
  const float* fc_b   = (const float*)d_in[11];
  float* out = (float*)d_out;

  char* ws = (char*)d_ws;
  size_t off = 0;
  auto carve = [&](size_t bytes) -> char* {
    char* p = ws + off;
    off = (off + bytes + 255) & ~(size_t)255;
    return p;
  };
  unsigned short* e_bf  = (unsigned short*)carve((size_t)MTOK * E_ * 2);
  unsigned short* wih0b = (unsigned short*)carve((size_t)G3H * E_ * 2);
  unsigned short* whh0b = (unsigned short*)carve((size_t)G3H * H_ * 2);
  unsigned short* wih1b = (unsigned short*)carve((size_t)G3H * H_ * 2);
  unsigned short* whh1b = (unsigned short*)carve((size_t)G3H * H_ * 2);
  unsigned short* fcwb  = (unsigned short*)carve((size_t)V_ * H_ * 2);
  float*          gx    = (float*)carve((size_t)MTOK * G3H * 4);   // reused L0/L1
  unsigned short* h1b   = (unsigned short*)carve((size_t)MTOK * H_ * 2);
  unsigned short* h2b   = (unsigned short*)carve((size_t)MTOK * H_ * 2);
  float*          hstf  = (float*)carve((size_t)B_ * H_ * 4);
  unsigned short* hstb  = (unsigned short*)carve((size_t)B_ * H_ * 2);
  unsigned int*   bar   = (unsigned int*)carve(256);

  auto cast = [&](const float* s, unsigned short* d, int n) {
    cast_bf16_kernel<<<(n + 255) / 256, 256, 0, stream>>>(s, d, n);
  };
  cast(w_ih0, wih0b, G3H * E_);
  cast(w_hh0, whh0b, G3H * H_);
  cast(w_ih1, wih1b, G3H * H_);
  cast(w_hh1, whh1b, G3H * H_);
  cast(fc_w,  fcwb,  V_ * H_);

  embed_bf16_kernel<<<(MTOK * E_ + 255) / 256, 256, 0, stream>>>(x, emb, e_bf);

  // gx0 = e @ w_ih0^T + b_ih0   [4096 x 3072], K=512  (gx re-read -> RT stores)
  gemm_bf16_f32_kernel<false><<<dim3(G3H / 128, MTOK / 128), 256, 0, stream>>>(
      e_bf, wih0b, b_ih0, gx, MTOK, G3H, E_);

  (void)hipMemsetAsync(hstf, 0, (size_t)B_ * H_ * 4, stream);
  (void)hipMemsetAsync(hstb, 0, (size_t)B_ * H_ * 2, stream);
  (void)hipMemsetAsync(bar,  0, 256, stream);
  gru_persistent_kernel<<<GRU_WGS, 96, 0, stream>>>(gx, whh0b, b_hh0, h1b,
                                                    hstf, hstb, bar);

  // gx1 = h1 @ w_ih1^T + b_ih1  [4096 x 3072], K=1024
  gemm_bf16_f32_kernel<false><<<dim3(G3H / 128, MTOK / 128), 256, 0, stream>>>(
      h1b, wih1b, b_ih1, gx, MTOK, G3H, H_);

  (void)hipMemsetAsync(hstf, 0, (size_t)B_ * H_ * 4, stream);
  (void)hipMemsetAsync(hstb, 0, (size_t)B_ * H_ * 2, stream);
  (void)hipMemsetAsync(bar,  0, 256, stream);
  gru_persistent_kernel<<<GRU_WGS, 96, 0, stream>>>(gx, whh1b, b_hh1, h2b,
                                                    hstf, hstb, bar);

  // logits = h2 @ fc_w^T + fc_b  [4096 x 32000], K=1024 -> dominant GEMM.
  // Write-once 524 MB output: stream with non-temporal stores.
  gemm_bf16_f32_kernel<true><<<dim3(V_ / 128, MTOK / 128), 256, 0, stream>>>(
      h2b, fcwb, fc_b, out, MTOK, V_, H_);
}
